// GATlayer_14551349199319
// MI455X (gfx1250) — matrix operand order repaired and checked by reading, not compiled, not run on hardware
//
#include <hip/hip_runtime.h>
#include <hip/hip_bf16.h>

typedef __attribute__((ext_vector_type(16))) __bf16 v16bf;
typedef __attribute__((ext_vector_type(8)))  __bf16 v8bf;
typedef __attribute__((ext_vector_type(8)))  float  v8f;

#define N_NODES 8192
#define D_IN    512
#define D_OUT   256

// Pin relative order of VMEM and WMMA; let VALU/SALU/DS/trans schedule freely.
__device__ __forceinline__ void sched_pin_mem_mma() {
#if __has_builtin(__builtin_amdgcn_sched_barrier)
    __builtin_amdgcn_sched_barrier(0x2 | 0x4 | 0x80 | 0x100 | 0x200 | 0x400);
#endif
}

// ---- fast math helpers (guarded: always compile, use native trans ops when available) ----
__device__ __forceinline__ float fastexp(float x) {   // e^x
#if __has_builtin(__builtin_amdgcn_exp2f)
    return __builtin_amdgcn_exp2f(x * 1.44269504088896340736f);
#else
    return __expf(x);
#endif
}
__device__ __forceinline__ float sigm_m05(float z) {  // sigmoid(z) - 0.5
#if __has_builtin(__builtin_amdgcn_tanhf)
    return 0.5f * __builtin_amdgcn_tanhf(0.5f * z);   // single v_tanh_f32
#else
    float e = fastexp(-z);
    return __builtin_amdgcn_rcpf(1.0f + e) - 0.5f;
#endif
}

// ---------------- Kernel 0: W [512,256] f32 -> WT [256,512] bf16 ----------------
__global__ void wtrans_kernel(const float* __restrict__ W, __bf16* __restrict__ WT) {
    int idx = blockIdx.x * 256 + threadIdx.x;   // 131072 total
    int k = idx >> 8;
    int n = idx & 255;
    WT[(size_t)n * D_IN + k] = (__bf16)W[idx];
}

// ---------------- Kernel 1: XT = (NF @ W)^T in bf16, via WMMA ----------------
__global__ __launch_bounds__(32) void xt_gemm_kernel(const float* __restrict__ NF,
                                                     const __bf16* __restrict__ WT,
                                                     __bf16* __restrict__ XT) {
    const int lane = threadIdx.x;
    const int bx = blockIdx.x;
    const int rt = bx >> 2, ng = bx & 3;
    const int rowbase = rt * 16;
    const int n  = lane & 15;
    const int g2 = lane >> 4;

    v8f acc[4];
    #pragma unroll
    for (int t = 0; t < 4; ++t) acc[t] = {};

    for (int kc = 0; kc < 16; ++kc) {
        const int kbase = kc * 32;
        const float* ap = NF + (size_t)(rowbase + n) * D_IN + kbase + g2 * 16;
        // issue ALL loads for this k-chunk up front, then pin them above the math
        float4 a0 = *(const float4*)(ap);
        float4 a1 = *(const float4*)(ap + 4);
        float4 b0 = *(const float4*)(ap + 8);
        float4 b1 = *(const float4*)(ap + 12);
        const __bf16* bp = WT + (size_t)(ng * 64 + n) * D_IN + kbase + g2 * 16;
        v16bf vb[4];
        #pragma unroll
        for (int t = 0; t < 4; ++t) vb[t] = *(const v16bf*)(bp + (size_t)t * 16 * D_IN);
        sched_pin_mem_mma();

        v16bf va;
        va[0]=(__bf16)a0.x; va[1]=(__bf16)a0.y; va[2]=(__bf16)a0.z; va[3]=(__bf16)a0.w;
        va[4]=(__bf16)a1.x; va[5]=(__bf16)a1.y; va[6]=(__bf16)a1.z; va[7]=(__bf16)a1.w;
        va[8]=(__bf16)b0.x; va[9]=(__bf16)b0.y; va[10]=(__bf16)b0.z; va[11]=(__bf16)b0.w;
        va[12]=(__bf16)b1.x; va[13]=(__bf16)b1.y; va[14]=(__bf16)b1.z; va[15]=(__bf16)b1.w;

        #pragma unroll
        for (int t = 0; t < 4; ++t) {
            acc[t] = __builtin_amdgcn_wmma_f32_16x16x32_bf16(
                false, va, false, vb[t], (short)0, acc[t], false, false);
        }
    }
    #pragma unroll
    for (int t = 0; t < 4; ++t) {
        const int nb = ng * 4 + t;
        v8bf pk;
        #pragma unroll
        for (int r = 0; r < 8; ++r) pk[r] = (__bf16)acc[t][r];
        *(v8bf*)(XT + (size_t)(nb * 16 + n) * N_NODES + rowbase + g2 * 8) = pk;
    }
}

// ---------------- Kernel 2: f0 = X@v0, f1 = X@v1 from XT ----------------
__global__ void fvec_kernel(const __bf16* __restrict__ XT,
                            const float* __restrict__ v0, const float* __restrict__ v1,
                            float* __restrict__ f0, float* __restrict__ f1) {
    int i = blockIdx.x * 256 + threadIdx.x;
    float s0 = 0.f, s1 = 0.f;
    for (int d = 0; d < D_OUT; ++d) {
        float x = (float)XT[(size_t)d * N_NODES + i];
        s0 += x * v0[d];
        s1 += x * v1[d];
    }
    f0[i] = s0; f1[i] = s1;
}

// ---------------- Kernel 3: fused masked-softmax attention + aggregation + ELU ----------------
#define SACC_STRIDE 258
__global__ __launch_bounds__(256, 1) void gat_main_kernel(const int* __restrict__ graph,
                                                          const __bf16* __restrict__ XT,
                                                          const float* __restrict__ f0,
                                                          const float* __restrict__ f1,
                                                          float* __restrict__ out) {
    const int tid  = threadIdx.x;
    const int lane = tid & 31;
    const int wave = tid >> 5;
    const int rowbase = blockIdx.x * 16;
    const int n  = lane & 15;
    const int g2 = lane >> 4;

    __shared__ float sacc[16 * SACC_STRIDE];
    __shared__ float sden[16];
    __shared__ float sf1[N_NODES];              // stage all of f1 (32KB) once per block
    for (int idx = tid; idx < 16 * SACC_STRIDE; idx += 256) sacc[idx] = 0.f;
    for (int idx = tid; idx < N_NODES; idx += 256) sf1[idx] = f1[idx];
    if (tid < 16) sden[tid] = 0.f;
    __syncthreads();

    const int i = rowbase + n;
    const float f0i = f0[i];
    const size_t grow = (size_t)i * N_NODES;
    const int segbase = wave * 1024;

    v8f acc[16];
    #pragma unroll
    for (int t = 0; t < 16; ++t) acc[t] = {};
    float dsum = 0.f;

    for (int c = 0; c < 32; ++c) {
        const int jA = segbase + c * 32 + g2 * 16;

        // prefetch graph (HBM stream) two chunks ahead; register-free
        if (c + 2 < 32) {
            __builtin_prefetch(graph + grow + jA + 64, 0, 1);
            __builtin_prefetch(graph + grow + jA + 80, 0, 1);
        }

        // issue first B tiles + graph loads early; pin above weight math so
        // their latency hides under the ~100-instr transcendental block
        const __bf16* bbase = XT + (size_t)n * N_NODES + (size_t)(segbase + c * 32) + g2 * 16;
        v16bf vb[4];
        #pragma unroll
        for (int p = 0; p < 4; ++p)
            vb[p] = *(const v16bf*)(bbase + (size_t)p * 16 * N_NODES);

        const int4 gA0 = *(const int4*)(graph + grow + jA);
        const int4 gA1 = *(const int4*)(graph + grow + jA + 4);
        const int4 gB0 = *(const int4*)(graph + grow + jA + 8);
        const int4 gB1 = *(const int4*)(graph + grow + jA + 12);
        sched_pin_mem_mma();

        const float4 fA0 = *(const float4*)(sf1 + jA);        // LDS broadcast reads
        const float4 fA1 = *(const float4*)(sf1 + jA + 4);
        const float4 fB0 = *(const float4*)(sf1 + jA + 8);
        const float4 fB1 = *(const float4*)(sf1 + jA + 12);

        v16bf va;
        auto wcalc = [&](int g, float fv) -> __bf16 {
            float wv = (g != 0) ? fastexp(sigm_m05(f0i + fv)) : 0.0f;
            dsum += wv;
            return (__bf16)wv;
        };
        va[0]=wcalc(gA0.x,fA0.x); va[1]=wcalc(gA0.y,fA0.y);
        va[2]=wcalc(gA0.z,fA0.z); va[3]=wcalc(gA0.w,fA0.w);
        va[4]=wcalc(gA1.x,fA1.x); va[5]=wcalc(gA1.y,fA1.y);
        va[6]=wcalc(gA1.z,fA1.z); va[7]=wcalc(gA1.w,fA1.w);
        va[8]=wcalc(gB0.x,fB0.x); va[9]=wcalc(gB0.y,fB0.y);
        va[10]=wcalc(gB0.z,fB0.z); va[11]=wcalc(gB0.w,fB0.w);
        va[12]=wcalc(gB1.x,fB1.x); va[13]=wcalc(gB1.y,fB1.y);
        va[14]=wcalc(gB1.z,fB1.z); va[15]=wcalc(gB1.w,fB1.w);

        // 16 WMMAs with a rotating 4-deep B-load pipeline; sched barriers keep
        // each load 4 WMMAs ahead of its consumer (4 distinct in-flight buffers)
        #pragma unroll
        for (int nb = 0; nb < 16; ++nb) {
            v16bf cur = vb[nb & 3];
            if (nb + 4 < 16)
                vb[nb & 3] = *(const v16bf*)(bbase + (size_t)(nb + 4) * 16 * N_NODES);
            sched_pin_mem_mma();
            acc[nb] = __builtin_amdgcn_wmma_f32_16x16x32_bf16(
                false, va, false, cur, (short)0, acc[nb], false, false);
        }
    }

    // cross-wave reduction in LDS (padded stride -> conflict-free ds_add_f32)
    #pragma unroll
    for (int nb = 0; nb < 16; ++nb) {
        #pragma unroll
        for (int r = 0; r < 8; ++r) {
            atomicAdd(&sacc[(r + 8 * g2) * SACC_STRIDE + nb * 16 + n], acc[nb][r]);
        }
    }
    atomicAdd(&sden[n], dsum);
    __syncthreads();

    for (int idx = tid; idx < 16 * D_OUT; idx += 256) {
        int mr = idx >> 8, d = idx & 255;
        float den = sden[mr];
        float v = (den > 0.f) ? sacc[mr * SACC_STRIDE + d] / den : 0.f;
        out[(size_t)(rowbase + mr) * D_OUT + d] = (v > 0.f) ? v : (fastexp(v) - 1.0f);
    }
}

extern "C" void kernel_launch(void* const* d_in, const int* in_sizes, int n_in,
                              void* d_out, int out_size, void* d_ws, size_t ws_size,
                              hipStream_t stream) {
    const float* NF    = (const float*)d_in[0];   // [8192,512] f32
    const int*   graph = (const int*)d_in[1];     // [8192,8192] i32
    const float* W     = (const float*)d_in[2];   // [512,256] f32
    const float* v0    = (const float*)d_in[3];   // [256] f32
    const float* v1    = (const float*)d_in[4];   // [256] f32
    float* out = (float*)d_out;                   // [8192,256] f32

    char* ws = (char*)d_ws;
    __bf16* WT = (__bf16*)ws;                                   // 256 KB
    __bf16* XT = (__bf16*)(ws + (256u << 10));                  // 4 MB
    float* f0 = (float*)(ws + (256u << 10) + (4u << 20));       // 32 KB
    float* f1 = f0 + N_NODES;                                   // 32 KB

    wtrans_kernel  <<<512, 256, 0, stream>>>(W, WT);
    xt_gemm_kernel <<<2048, 32, 0, stream>>>(NF, WT, XT);
    fvec_kernel    <<<32, 256, 0, stream>>>(XT, v0, v1, f0, f1);
    gat_main_kernel<<<512, 256, 0, stream>>>(graph, XT, f0, f1, out);
}
